// Adapter_60653528154703
// MI455X (gfx1250) — matrix operand order repaired and checked by reading, not compiled, not run on hardware
//
#include <hip/hip_runtime.h>
#include <hip/hip_bf16.h>

// ---------------- constants (match reference) ----------------
#define NN    20000
#define E1C   100000
#define E2C   100000
#define HIDC  128
#define NRBFC 64
#define CUTF  6.0f

typedef __attribute__((ext_vector_type(16))) __bf16 bf16x16;
typedef __attribute__((ext_vector_type(8)))  float  f32x8;

union ABfrag { bf16x16 v; uint4 q[2]; };

// WMMA wrapper: D = A(16x32 bf16) x B(32x16 bf16) + C(16x16 f32)
__device__ __forceinline__ f32x8 wmma_bf16(bf16x16 a, bf16x16 b, f32x8 c) {
    return __builtin_amdgcn_wmma_f32_16x16x32_bf16(false, a, false, b, (short)0, c, false, false);
}

// A-fragment from LDS tile s[16][LDK] (row-major, M x K), contiguous-k A layout (not the ISA table order):
// lane l: m=l%16, g=l/16; elems 0..15 = K[g*16 .. g*16+15] (contiguous, matches B)
template<int LDK>
__device__ __forceinline__ bf16x16 load_a_lds(const __bf16 (*s)[LDK], int kstep, int lane) {
    int m = lane & 15, g = lane >> 4;
    ABfrag r;
    r.q[0] = *(const uint4*)&s[m][kstep * 32 + g * 16];
    r.q[1] = *(const uint4*)&s[m][kstep * 32 + g * 16 + 8];
    return r.v;
}

// B-fragment from global row-major weight W[out_dim][ldk] (B = W^T):
// lane l: n=l%16, g=l/16; 16 contiguous K at kstep*32 + g*16
__device__ __forceinline__ bf16x16 load_b_glob(const __bf16* W, int ldk, int ntile, int kstep, int lane) {
    int n = lane & 15, g = lane >> 4;
    const uint4* p = (const uint4*)(W + (size_t)(ntile * 16 + n) * ldk + kstep * 32 + g * 16);
    ABfrag r;
    r.q[0] = p[0];
    r.q[1] = p[1];
    return r.v;
}

__device__ __forceinline__ float rbf_val(float d, int j) {
    float c = CUTF * (float)j * (1.0f / 63.0f);
    const float w = CUTF / 64.0f;
    float t = d * (1.0f / CUTF);
    t = fminf(fmaxf(t, 0.0f), 1.0f);
    float env = 0.5f * (__cosf(3.14159265358979f * t) + 1.0f);
    float diff = d - c;
    return __expf(-diff * diff * (1.0f / (2.0f * w * w))) * env;
}

// ---------------- K1: weight prep (bf16 convert + folded GEMMs) ----------------
// segments: wi 32768 | w1a 16384 | w2 49152 | wmsg 24576 | w2d 8192 | wrbf 8192 | bclass 384
__global__ void prep_weights_kernel(const float* __restrict__ Wi, const float* __restrict__ Wphi1,
                                    const float* __restrict__ Wphi2, const float* __restrict__ Wmsg,
                                    const float* __restrict__ We2d, const float* __restrict__ Wrbf,
                                    const float* __restrict__ bphi1, const float* __restrict__ etemb,
                                    __bf16* wi_bf, __bf16* w1a_bf, __bf16* w2_bf, __bf16* wmsg_bf,
                                    __bf16* w2d_bf, __bf16* wrbf_bf, float* b_class) {
    int t = blockIdx.x * 256 + threadIdx.x;
    if (t < 32768) { wi_bf[t] = (__bf16)Wi[t]; return; }
    t -= 32768;
    if (t < 16384) { int n = t >> 7, k = t & 127; w1a_bf[t] = (__bf16)Wphi1[n * 288 + k]; return; }
    t -= 16384;
    if (t < 49152) { w2_bf[t] = (__bf16)Wphi2[t]; return; }
    t -= 49152;
    if (t < 24576) { wmsg_bf[t] = (__bf16)Wmsg[t]; return; }
    t -= 24576;
    if (t < 8192) {   // w2d_comb[n][k] = sum_f W1b[n][f] * We2d[f][k]
        int n = t >> 6, k = t & 63;
        float s = 0.0f;
        for (int f = 0; f < 128; ++f) s += Wphi1[n * 288 + 128 + f] * We2d[f * 64 + k];
        w2d_bf[t] = (__bf16)s; return;
    }
    t -= 8192;
    if (t < 8192) {   // wrbf_comb[n][k] = sum_f W1b[n][f] * Wrbf[f][k]
        int n = t >> 6, k = t & 63;
        float s = 0.0f;
        for (int f = 0; f < 128; ++f) s += Wphi1[n * 288 + 128 + f] * Wrbf[f * 64 + k];
        wrbf_bf[t] = (__bf16)s; return;
    }
    t -= 8192;
    if (t < 384) {    // b_class[c][n] = bphi1[n] + sum_j W1c[n][j] * et[c][j]
        int c = t >> 7, n = t & 127;
        float s = bphi1[n];
        for (int j = 0; j < 32; ++j) s += Wphi1[n * 288 + 256 + j] * etemb[c * 32 + j];
        b_class[t] = s;
    }
}

// ---------------- K2: mask_dist scatter ----------------
__global__ void mask_scatter_kernel(const int* __restrict__ Edist_idx, unsigned int* mask_dist) {
    int t = blockIdx.x * 256 + threadIdx.x;
    if (t < 2 * E2C) atomicOr(&mask_dist[Edist_idx[t]], 1u);
}

// ---------------- K3: node projection nodes = [H|H2d or virt] @ Wi^T -> bf16 ----------------
__global__ void node_proj_kernel(const float* __restrict__ H, const float* __restrict__ H2d,
                                 const float* __restrict__ virt, const __bf16* __restrict__ wi_bf,
                                 __bf16* __restrict__ nodes_bf) {
    __shared__ __align__(16) __bf16 s_in[16][256];
    int tid = threadIdx.x;            // 128 threads = 4 waves
    int base = blockIdx.x * 16;       // row tile over 2N rows
    for (int i = tid; i < 16 * 256; i += 128) {
        int r = i >> 8, c = i & 255;
        int R = base + r;
        float val;
        if (c < 128) val = (R < NN) ? H[(size_t)R * 128 + c] : H[(size_t)(R - NN) * 128 + c];
        else         val = (R < NN) ? H2d[(size_t)R * 128 + (c - 128)] : virt[c - 128];
        s_in[r][c] = (__bf16)val;
    }
    __syncthreads();
    int wave = tid >> 5, lane = tid & 31;
    int ln = lane & 15, g = lane >> 4;
    for (int nt = wave * 2; nt < wave * 2 + 2; ++nt) {
        f32x8 acc = {};
#pragma unroll
        for (int ks = 0; ks < 8; ++ks)
            acc = wmma_bf16(load_a_lds<256>(s_in, ks, lane), load_b_glob(wi_bf, 256, nt, ks, lane), acc);
        int ncol = nt * 16 + ln;
#pragma unroll
        for (int v = 0; v < 8; ++v) {
            int R = base + g * 8 + v;
            nodes_bf[(size_t)R * 128 + ncol] = (__bf16)acc[v];
        }
    }
}

// ---------------- K4: fused edge kernel (16 edges / workgroup, 4 waves) ----------------
__global__ void __launch_bounds__(128)
edge_kernel(const float* __restrict__ Zr, const float* __restrict__ Z3, const float* __restrict__ Vin,
            const int* __restrict__ E2d_idx, const float* __restrict__ E2d_feat,
            const int* __restrict__ Edist_idx, const float* __restrict__ Edist_val,
            const __bf16* __restrict__ nodes_bf,
            const __bf16* __restrict__ w1a, const __bf16* __restrict__ w2d_c,
            const __bf16* __restrict__ wrbf_c, const __bf16* __restrict__ w2,
            const __bf16* __restrict__ wmsg,
            const float* __restrict__ b_class, const float* __restrict__ b2, const float* __restrict__ bm,
            float* outH, float* outV) {
    __shared__ int   s_row[16];
    __shared__ int   s_col[16];
    __shared__ float s_unit[16][3];
    __shared__ __align__(16) __bf16 s_rbfd[16][64];
    __shared__ __align__(16) __bf16 s_feat[16][64];
    __shared__ __align__(16) __bf16 s_nodes[16][128];
    __shared__ __align__(16) __bf16 s_hidden[16][128];
    __shared__ __align__(16) float  s_m[16][384];

    int tid = threadIdx.x;
    int tb  = blockIdx.x * 16;                       // compact edge tile base (never straddles classes)
    int cls = (tb < E1C) ? 0 : ((tb < E1C + NN) ? 1 : 2);

    int el = tid >> 3;      // edge-in-tile 0..15
    int sub = tid & 7;      // 8 threads per edge
    int e = tb + el;

    int row, col; float dval = 0.0f;
    if (cls == 0)      { row = E2d_idx[e]; col = E2d_idx[E1C + e]; }
    else if (cls == 1) { int i = e - E1C; row = i; col = i + NN; }  // only kept half of E3d
    else               { int j = e - E1C - NN; row = Edist_idx[j]; col = Edist_idx[E2C + j]; dval = Edist_val[j]; }

    float zr0, zr1, zr2, zc0, zc1, zc2;
    { const float* p = (row < NN) ? &Zr[(size_t)row * 3] : &Z3[(size_t)(row - NN) * 3];
      zr0 = p[0]; zr1 = p[1]; zr2 = p[2]; }
    { const float* p = (col < NN) ? &Zr[(size_t)col * 3] : &Z3[(size_t)(col - NN) * 3];
      zc0 = p[0]; zc1 = p[1]; zc2 = p[2]; }
    float r0 = zr0 - zc0, r1 = zr1 - zc1, r2 = zr2 - zc2;
    float dist = sqrtf(r0 * r0 + r1 * r1 + r2 * r2 + 1e-8f);
    float inv = 1.0f / dist;
    if (sub == 0) {
        s_row[el] = row; s_col[el] = col;
        s_unit[el][0] = r0 * inv; s_unit[el][1] = r1 * inv; s_unit[el][2] = r2 * inv;
    }
#pragma unroll
    for (int jj = 0; jj < 8; ++jj) {
        int j = sub * 8 + jj;
        s_rbfd[el][j] = (__bf16)rbf_val(dist, j);
        if (cls == 0)      s_feat[el][j] = (__bf16)E2d_feat[(size_t)j * E1C + e];
        else if (cls == 2) s_feat[el][j] = (__bf16)rbf_val(dval, j);
    }
    {   // gather nodes_bf[col] row into LDS: 16 bf16 (32B) per thread
        const uint4* src = (const uint4*)(nodes_bf + (size_t)col * 128 + sub * 16);
        uint4* dst = (uint4*)(&s_nodes[el][sub * 16]);
        dst[0] = src[0]; dst[1] = src[1];
    }
    __syncthreads();

    int wave = tid >> 5, lane = tid & 31;
    int ln = lane & 15, g = lane >> 4;

    // GEMM1: hidden = silu(nodes[col] @ W1a^T + feat @ Wc^T + b_class)
    for (int nt = wave * 2; nt < wave * 2 + 2; ++nt) {
        f32x8 acc = {};
#pragma unroll
        for (int ks = 0; ks < 4; ++ks)
            acc = wmma_bf16(load_a_lds<128>(s_nodes, ks, lane), load_b_glob(w1a, 128, nt, ks, lane), acc);
        if (cls != 1) {
            const __bf16* wc = (cls == 0) ? w2d_c : wrbf_c;
#pragma unroll
            for (int ks = 0; ks < 2; ++ks)
                acc = wmma_bf16(load_a_lds<64>(s_feat, ks, lane), load_b_glob(wc, 64, nt, ks, lane), acc);
        }
        int ncol = nt * 16 + ln;
        float bias = b_class[cls * 128 + ncol];
#pragma unroll
        for (int v = 0; v < 8; ++v) {
            float x = acc[v] + bias;
            float h = x / (1.0f + __expf(-x));           // silu
            s_hidden[g * 8 + v][ncol] = (__bf16)h;
        }
    }
    __syncthreads();

    // GEMM2: m = (hidden @ W2^T + b2) * (rbf(dist) @ Wmsg^T + bmsg)
    for (int nt = wave * 6; nt < wave * 6 + 6; ++nt) {
        f32x8 acc = {}, acc2 = {};
#pragma unroll
        for (int ks = 0; ks < 4; ++ks)
            acc = wmma_bf16(load_a_lds<128>(s_hidden, ks, lane), load_b_glob(w2, 128, nt, ks, lane), acc);
#pragma unroll
        for (int ks = 0; ks < 2; ++ks)
            acc2 = wmma_bf16(load_a_lds<64>(s_rbfd, ks, lane), load_b_glob(wmsg, 64, nt, ks, lane), acc2);
        int ncol = nt * 16 + ln;
        float bb2 = b2[ncol], bbm = bm[ncol];
#pragma unroll
        for (int v = 0; v < 8; ++v)
            s_m[g * 8 + v][ncol] = (acc[v] + bb2) * (acc2[v] + bbm);
    }
    __syncthreads();

    // Fused scatter: H_add += s_msg ; V_add += v[col]*v_gate + unit*r_gate (combined -> 1 atomic per elem)
    int re = s_row[el];                  // always < NN for processed edges
    int ce = s_col[el];
    float u0 = s_unit[el][0], u1 = s_unit[el][1], u2 = s_unit[el][2];
    float* hdst = outH + (size_t)re * 128;
    float* vdst = outV + (size_t)re * 384;
    const float* vsrc = Vin + (size_t)(cls == 1 ? 0 : ce) * 384;
#pragma unroll 4
    for (int kk = 0; kk < 16; ++kk) {
        int k = sub * 16 + kk;
        atomicAdd(&hdst[k], s_m[el][k]);
        float vg = s_m[el][128 + k];
        float rg = s_m[el][256 + k];
        if (cls == 1) {                  // v[col] == 0 for virtual nodes
            atomicAdd(&vdst[k * 3 + 0], rg * u0);
            atomicAdd(&vdst[k * 3 + 1], rg * u1);
            atomicAdd(&vdst[k * 3 + 2], rg * u2);
        } else {
            atomicAdd(&vdst[k * 3 + 0], vg * vsrc[k * 3 + 0] + rg * u0);
            atomicAdd(&vdst[k * 3 + 1], vg * vsrc[k * 3 + 1] + rg * u1);
            atomicAdd(&vdst[k * 3 + 2], vg * vsrc[k * 3 + 2] + rg * u2);
        }
    }
}

// ---------------- K5: clip + mask finalize ----------------
__global__ void finalize_kernel(float* outH, float* outV, const unsigned char* __restrict__ m2d,
                                const unsigned char* __restrict__ m3d,
                                const unsigned int* __restrict__ mdist) {
    int i = blockIdx.x * 256 + threadIdx.x;      // over N*128
    if (i >= NN * 128) return;
    int node = i >> 7;
    bool msk = (m2d[node] | m3d[node]) || (mdist[node] != 0u);
    float h = outH[i];
    outH[i] = msk ? fminf(fmaxf(h, -100.0f), 100.0f) : 0.0f;
#pragma unroll
    for (int d = 0; d < 3; ++d) {
        float v = outV[(size_t)i * 3 + d];
        outV[(size_t)i * 3 + d] = msk ? fminf(fmaxf(v, -100.0f), 100.0f) : 0.0f;
    }
}

extern "C" void kernel_launch(void* const* d_in, const int* in_sizes, int n_in,
                              void* d_out, int out_size, void* d_ws, size_t ws_size,
                              hipStream_t stream) {
    const float* H      = (const float*)d_in[0];
    const float* V      = (const float*)d_in[1];
    const float* Z      = (const float*)d_in[2];
    const float* H2d    = (const float*)d_in[4];
    const unsigned char* mask2d = (const unsigned char*)d_in[5];
    const int*   E2didx = (const int*)d_in[6];
    const float* E2dfeat= (const float*)d_in[7];
    const float* Z3     = (const float*)d_in[8];
    const unsigned char* mask3d = (const unsigned char*)d_in[9];
    const int*   Edidx  = (const int*)d_in[10];
    const float* Edval  = (const float*)d_in[11];
    const float* virt   = (const float*)d_in[12];
    const float* etemb  = (const float*)d_in[13];
    const float* Wrbf   = (const float*)d_in[14];
    const float* We2d   = (const float*)d_in[15];
    const float* Wi     = (const float*)d_in[16];
    const float* Wphi1  = (const float*)d_in[17];
    const float* bphi1  = (const float*)d_in[18];
    const float* Wphi2  = (const float*)d_in[19];
    const float* bphi2  = (const float*)d_in[20];
    const float* Wmsg   = (const float*)d_in[21];
    const float* bmsg   = (const float*)d_in[22];

    size_t off = 0;
    auto carve = [&](size_t bytes) -> void* {
        void* r = (char*)d_ws + off;
        off += (bytes + 255) & ~(size_t)255;
        return r;
    };
    __bf16* nodes_bf = (__bf16*)carve((size_t)2 * NN * 128 * 2);
    __bf16* wi_bf    = (__bf16*)carve(128 * 256 * 2);
    __bf16* w1a_bf   = (__bf16*)carve(128 * 128 * 2);
    __bf16* w2_bf    = (__bf16*)carve(384 * 128 * 2);
    __bf16* wmsg_bf  = (__bf16*)carve(384 * 64 * 2);
    __bf16* w2d_bf   = (__bf16*)carve(128 * 64 * 2);
    __bf16* wrbf_bf  = (__bf16*)carve(128 * 64 * 2);
    float*  b_class  = (float*)carve(3 * 128 * 4);
    unsigned int* mask_dist = (unsigned int*)carve((size_t)NN * 4);

    float* outH = (float*)d_out;
    float* outV = outH + (size_t)NN * 128;

    hipMemsetAsync(d_out, 0, (size_t)out_size * sizeof(float), stream);
    hipMemsetAsync(mask_dist, 0, (size_t)NN * 4, stream);

    prep_weights_kernel<<<546, 256, 0, stream>>>(Wi, Wphi1, Wphi2, Wmsg, We2d, Wrbf, bphi1, etemb,
                                                 wi_bf, w1a_bf, w2_bf, wmsg_bf, w2d_bf, wrbf_bf, b_class);
    mask_scatter_kernel<<<(2 * E2C + 255) / 256, 256, 0, stream>>>(Edidx, mask_dist);
    node_proj_kernel<<<(2 * NN) / 16, 128, 0, stream>>>(H, H2d, virt, wi_bf, nodes_bf);
    edge_kernel<<<(E1C + NN + E2C) / 16, 128, 0, stream>>>(Z, Z3, V, E2didx, E2dfeat, Edidx, Edval,
                                                           nodes_bf, w1a_bf, w2d_bf, wrbf_bf, w2_bf,
                                                           wmsg_bf, b_class, bphi2, bmsg, outH, outV);
    finalize_kernel<<<(NN * 128 + 255) / 256, 256, 0, stream>>>(outH, outV, mask2d, mask3d, mask_dist);
}
